// ImageCPPN_31980326486352
// MI455X (gfx1250) — matrix operand order repaired and checked by reading, not compiled, not run on hardware
//
#include <hip/hip_runtime.h>
#include <math.h>

#define NB   8
#define NHID 64
#define PIX  (512*512)
#define HROW 72            // LDS row stride in bf16 (64 + 8 pad, keeps 16B align)
#define EPSV 1e-5f

typedef __bf16 bf16_t;
typedef __attribute__((ext_vector_type(16))) __bf16 v16bf;
typedef __attribute__((ext_vector_type(8)))  float  v8f;

// ---------- bf16 <-> f32 (RNE) ----------
static __device__ __forceinline__ bf16_t f2bf(float f) {
  union { float f; unsigned u; } a; a.f = f;
  unsigned r = a.u + 0x7FFFu + ((a.u >> 16) & 1u);
  union { unsigned short s; bf16_t b; } o; o.s = (unsigned short)(r >> 16);
  return o.b;
}
static __device__ __forceinline__ float bf2f(bf16_t b) {
  union { unsigned short s; bf16_t b; } i; i.b = b;
  union { unsigned u; float f; } o; o.u = ((unsigned)i.s) << 16;
  return o.f;
}

// ---------- fragment loaders (LDS -> VGPR, per CDNA5 16-bit WMMA layouts) ----------
// B (32x16, KxN): lane<16 -> K {0..7,16..23}; lane>=16 -> K {8..15,24..31}; col N = lane&15.
static __device__ __forceinline__ v16bf loadB(const bf16_t* sHrow, int k0) {
  union { uint4 q[2]; v16bf v; } u;
  const uint4* src = reinterpret_cast<const uint4*>(sHrow + k0);
  u.q[0] = src[0]; u.q[1] = src[2];
  return u.v;
}
// A (16x32, MxK): lane<16 -> K {0..7,16..23}; lane>=16 -> K {8..15,24..31}; row M = lane&15.
static __device__ __forceinline__ v16bf loadA(const bf16_t* sWl, int mt, int kt, int lane) {
  int m = lane & 15;
  const bf16_t* row = sWl + (mt * 16 + m) * 64;
  int base = kt * 32 + ((lane & 16) ? 8 : 0);
  union { uint4 q[2]; v16bf v; } u;
  u.q[0] = *reinterpret_cast<const uint4*>(row + base);
  u.q[1] = *reinterpret_cast<const uint4*>(row + base + 16);
  return u.v;
}

// ---------- one 64x64 x 16px layer: relu(W*h + b), optional residual / LN stats ----------
template <bool RESID, bool STATS>
static __device__ __forceinline__ void gemm_layer(
    const bf16_t* sWl, const float* sBl,
    const bf16_t* sHin, bf16_t* sOut,
    float* sSum, float* sSq, int lane, int pl) {
  const int hi8 = (lane & 16) ? 8 : 0;
  v8f acc[4];
#pragma unroll
  for (int mt = 0; mt < 4; ++mt)
#pragma unroll
    for (int r = 0; r < 8; ++r) acc[mt][r] = sBl[mt * 16 + r + hi8];

#pragma unroll
  for (int kt = 0; kt < 2; ++kt) {
    v16bf bfrag = loadB(sHin + pl * HROW, kt * 32 + ((lane & 16) ? 8 : 0));
#pragma unroll
    for (int mt = 0; mt < 4; ++mt) {
      v16bf afrag = loadA(sWl, mt, kt, lane);
      acc[mt] = __builtin_amdgcn_wmma_f32_16x16x32_bf16(
          false, afrag, false, bfrag, (short)0, acc[mt], false, false);
    }
  }

#pragma unroll
  for (int mt = 0; mt < 4; ++mt) {
#pragma unroll
    for (int r = 0; r < 8; ++r) {
      int o = mt * 16 + r + hi8;
      float lx = fmaxf(acc[mt][r], 0.0f);
      if (STATS) {
        float s = lx, q = lx * lx;
#pragma unroll
        for (int m = 8; m >= 1; m >>= 1) {
          s += __shfl_xor(s, m, 32);
          q += __shfl_xor(q, m, 32);
        }
        if ((lane & 15) == 0) { atomicAdd(&sSum[o], s); atomicAdd(&sSq[o], q); }
      }
      float v = lx;
      if (RESID) v += bf2f(sHin[pl * HROW + o]);
      sOut[pl * HROW + o] = f2bf(v);
    }
  }
}

// ---------- K0: input layer + hidden layers 1..5 (residual, no LN) + layer-6 matmul + stats ----------
__global__ void __launch_bounds__(256) k_front(
    const float* __restrict__ x, const float* __restrict__ Win, const float* __restrict__ bin,
    const float* __restrict__ Whid, const float* __restrict__ bhid,
    bf16_t* __restrict__ hbuf, bf16_t* __restrict__ lxbuf,
    float* __restrict__ gsum, float* __restrict__ gsq) {
  __shared__ __align__(16) bf16_t sW[6 * 4096];
  __shared__ float sWin[192], sBin[64], sBh[6 * 64];
  __shared__ __align__(16) bf16_t sH[128 * HROW];
  __shared__ __align__(16) bf16_t sLx[128 * HROW];
  __shared__ float sSum[64], sSq[64];

  const int tid = threadIdx.x;
  const int b = blockIdx.x >> 6, blk = blockIdx.x & 63;

  for (int i = tid; i < 6 * 4096; i += 256) {
    int jj = i >> 12, rem = i & 4095;
    sW[i] = f2bf(Whid[((size_t)(jj * NB + b)) * 4096 + rem]);
  }
  if (tid < 192) sWin[tid] = Win[b * 192 + tid];
  if (tid < 64) { sBin[tid] = bin[b * 64 + tid]; sSum[tid] = 0.f; sSq[tid] = 0.f; }
  for (int i = tid; i < 6 * 64; i += 256) {
    int jj = i >> 6, c = i & 63;
    sBh[i] = bhid[(jj * NB + b) * 64 + c];
  }
  __syncthreads();

  const int lane = tid & 31, wv = tid >> 5;
  const int pl = wv * 16 + (lane & 15);
  const size_t baseP = (size_t)blk * 4096;

  for (int chunk = 0; chunk < 32; ++chunk) {
    const size_t pb = baseP + (size_t)chunk * 128;
    // input layer (K=3, scalar FMAs)
    const size_t pg = pb + pl;
    float x0 = x[((size_t)(b * 3 + 0)) * PIX + pg];
    float x1 = x[((size_t)(b * 3 + 1)) * PIX + pg];
    float x2 = x[((size_t)(b * 3 + 2)) * PIX + pg];
    const int ob = (lane & 16) ? 32 : 0;
#pragma unroll
    for (int t = 0; t < 32; ++t) {
      int o = ob + t;
      float v = fmaf(sWin[o * 3 + 2], x2, fmaf(sWin[o * 3 + 1], x1, fmaf(sWin[o * 3], x0, sBin[o])));
      sH[pl * HROW + o] = f2bf(fmaxf(v, 0.f));
    }
    // hidden layers 1..5: h = h + relu(W h + b)
#pragma unroll
    for (int j = 0; j < 5; ++j)
      gemm_layer<true, false>(sW + j * 4096, sBh + j * 64, sH, sH, nullptr, nullptr, lane, pl);
    // layer 6 matmul: lx6 = relu(W5 h + b5), accumulate LN stats
    gemm_layer<false, true>(sW + 5 * 4096, sBh + 5 * 64, sH, sLx, sSum, sSq, lane, pl);
    __syncthreads();
    for (int i = tid; i < 8192; i += 256) {
      int p = i & 127, c = i >> 7;
      size_t g = ((size_t)(b * 64 + c)) * PIX + pb + p;
      hbuf[g]  = sH[p * HROW + c];
      lxbuf[g] = sLx[p * HROW + c];
    }
    __syncthreads();
  }
  if (tid < 64)       atomicAdd(&gsum[b * 64 + tid], sSum[tid]);
  else if (tid < 128) atomicAdd(&gsq[b * 64 + tid - 64], sSq[tid - 64]);
}

// ---------- K1: finalize LN stats, re-zero accumulators ----------
__global__ void k_finalize(float* gsum, float* gsq, float* gmu, float* grs) {
  int i = threadIdx.x;  // 512 = B*NHID
  float s = gsum[i], q = gsq[i];
  float mean = s * (1.0f / (float)PIX);
  float var  = q * (1.0f / (float)PIX) - mean * mean;
  gmu[i] = mean;
  grs[i] = rsqrtf(var + EPSV);
  gsum[i] = 0.f;
  gsq[i]  = 0.f;
}

// ---------- K2: h += norm(lx); next lx = relu(W h + b); stats ----------
__global__ void __launch_bounds__(256) k_mid(
    const float* __restrict__ Whid, const float* __restrict__ bhid,
    const float* __restrict__ gmu, const float* __restrict__ grs,
    bf16_t* __restrict__ hbuf, bf16_t* __restrict__ lxbuf,
    float* __restrict__ gsum, float* __restrict__ gsq, int j) {
  __shared__ __align__(16) bf16_t sW[4096];
  __shared__ float sBh[64], sMu[64], sRs[64];
  __shared__ __align__(16) bf16_t sH[128 * HROW];
  __shared__ __align__(16) bf16_t sLx[128 * HROW];
  __shared__ float sSum[64], sSq[64];

  const int tid = threadIdx.x;
  const int b = blockIdx.x >> 6, blk = blockIdx.x & 63;

  for (int i = tid; i < 4096; i += 256)
    sW[i] = f2bf(Whid[((size_t)(j * NB + b)) * 4096 + i]);
  if (tid < 64) {
    sBh[tid] = bhid[(j * NB + b) * 64 + tid];
    sMu[tid] = gmu[b * 64 + tid];
    sRs[tid] = grs[b * 64 + tid];
    sSum[tid] = 0.f; sSq[tid] = 0.f;
  }
  __syncthreads();

  const int lane = tid & 31, wv = tid >> 5;
  const int pl = wv * 16 + (lane & 15);
  const size_t baseP = (size_t)blk * 4096;

  for (int chunk = 0; chunk < 32; ++chunk) {
    const size_t pb = baseP + (size_t)chunk * 128;
    for (int i = tid; i < 8192; i += 256) {
      int p = i & 127, c = i >> 7;
      size_t g = ((size_t)(b * 64 + c)) * PIX + pb + p;
      float hn = bf2f(hbuf[g]) + (bf2f(lxbuf[g]) - sMu[c]) * sRs[c];
      bf16_t hb = f2bf(hn);
      hbuf[g] = hb;
      sH[p * HROW + c] = hb;
    }
    __syncthreads();
    gemm_layer<false, true>(sW, sBh, sH, sLx, sSum, sSq, lane, pl);
    __syncthreads();
    for (int i = tid; i < 8192; i += 256) {
      int p = i & 127, c = i >> 7;
      lxbuf[((size_t)(b * 64 + c)) * PIX + pb + p] = sLx[p * HROW + c];
    }
    __syncthreads();
  }
  if (tid < 64)       atomicAdd(&gsum[b * 64 + tid], sSum[tid]);
  else if (tid < 128) atomicAdd(&gsq[b * 64 + tid - 64], sSq[tid - 64]);
}

// ---------- K3: h += norm(lx); out = sigmoid(Wout h + bout) ----------
__global__ void __launch_bounds__(256) k_out(
    const float* __restrict__ Wout, const float* __restrict__ bout,
    const float* __restrict__ gmu, const float* __restrict__ grs,
    const bf16_t* __restrict__ hbuf, const bf16_t* __restrict__ lxbuf,
    float* __restrict__ out) {
  __shared__ float sWo[192], sBo[3], sMu[64], sRs[64];
  __shared__ __align__(16) bf16_t sH[128 * HROW];

  const int tid = threadIdx.x;
  const int b = blockIdx.x >> 6, blk = blockIdx.x & 63;

  if (tid < 192) sWo[tid] = Wout[b * 192 + tid];
  if (tid < 3)   sBo[tid] = bout[b * 3 + tid];
  if (tid < 64)  { sMu[tid] = gmu[b * 64 + tid]; sRs[tid] = grs[b * 64 + tid]; }
  __syncthreads();

  const size_t baseP = (size_t)blk * 4096;
  for (int chunk = 0; chunk < 32; ++chunk) {
    const size_t pb = baseP + (size_t)chunk * 128;
    for (int i = tid; i < 8192; i += 256) {
      int p = i & 127, c = i >> 7;
      size_t g = ((size_t)(b * 64 + c)) * PIX + pb + p;
      float hn = bf2f(hbuf[g]) + (bf2f(lxbuf[g]) - sMu[c]) * sRs[c];
      sH[p * HROW + c] = f2bf(hn);
    }
    __syncthreads();
    if (tid < 128) {
      int p = tid;
      float z0 = sBo[0], z1 = sBo[1], z2 = sBo[2];
      for (int o = 0; o < 64; ++o) {
        float hv = bf2f(sH[p * HROW + o]);
        z0 = fmaf(sWo[o], hv, z0);
        z1 = fmaf(sWo[64 + o], hv, z1);
        z2 = fmaf(sWo[128 + o], hv, z2);
      }
      out[((size_t)(b * 3 + 0)) * PIX + pb + p] = 1.f / (1.f + expf(-z0));
      out[((size_t)(b * 3 + 1)) * PIX + pb + p] = 1.f / (1.f + expf(-z1));
      out[((size_t)(b * 3 + 2)) * PIX + pb + p] = 1.f / (1.f + expf(-z2));
    }
    __syncthreads();
  }
}

__global__ void k_zero(float* p, int n) {
  int i = blockIdx.x * 256 + threadIdx.x;
  if (i < n) p[i] = 0.f;
}

extern "C" void kernel_launch(void* const* d_in, const int* in_sizes, int n_in,
                              void* d_out, int out_size, void* d_ws, size_t ws_size,
                              hipStream_t stream) {
  const float* x    = (const float*)d_in[0];
  const float* Win  = (const float*)d_in[1];
  const float* bin  = (const float*)d_in[2];
  const float* Whid = (const float*)d_in[3];
  const float* bhid = (const float*)d_in[4];
  const float* Wout = (const float*)d_in[5];
  const float* bout = (const float*)d_in[6];
  float* out = (float*)d_out;

  const size_t actElems = (size_t)NB * NHID * PIX;        // 134,217,728
  bf16_t* hbuf  = (bf16_t*)d_ws;
  bf16_t* lxbuf = hbuf + actElems;
  float*  stats = (float*)((char*)d_ws + 2 * actElems * sizeof(bf16_t));
  float*  gsum = stats;
  float*  gsq  = stats + 512;
  float*  gmu  = stats + 1024;
  float*  grs  = stats + 1536;

  k_zero<<<4, 256, 0, stream>>>(gsum, 1024);  // gsum + gsq contiguous
  k_front<<<512, 256, 0, stream>>>(x, Win, bin, Whid, bhid, hbuf, lxbuf, gsum, gsq);
  for (int j = 6; j <= 10; ++j) {
    k_finalize<<<1, 512, 0, stream>>>(gsum, gsq, gmu, grs);
    k_mid<<<512, 256, 0, stream>>>(Whid, bhid, gmu, grs, hbuf, lxbuf, gsum, gsq, j);
  }
  k_finalize<<<1, 512, 0, stream>>>(gsum, gsq, gmu, grs);
  k_out<<<512, 256, 0, stream>>>(Wout, bout, gmu, grs, hbuf, lxbuf, out);
}
